// AttentionDecoder_29033978921339
// MI455X (gfx1250) — matrix operand order repaired and checked by reading, not compiled, not run on hardware
//
#include <hip/hip_runtime.h>

// ---------------- types ----------------
typedef __attribute__((ext_vector_type(8)))  float  floatx8;
typedef __attribute__((ext_vector_type(8)))  float  f32x8;
typedef __attribute__((ext_vector_type(8)))  __bf16 bf16x8;
typedef __attribute__((ext_vector_type(16))) __bf16 bf16x16;

// ---------------- problem sizes ----------------
#define NB   32      // batch
#define TD   64      // decoder steps
#define TE   64      // encoder len
#define NV   32000   // vocab
#define ND   512     // embed dim
#define NH   512     // hidden
#define G4H  2048    // 4*H
#define KF   1024    // 2*H (feat K)
#define NM   (NB*TD) // 2048 rows

// d_out layout (floats): logits | hT | cT | attn
#define HT_OFF  ((size_t)NM * NV)                 // 65,536,000
#define CT_OFF  (HT_OFF + (size_t)NB * NH)        // +16384
#define ATT_OFF (CT_OFF + (size_t)NB * NH)        // +16384

// d_ws layout (bytes)
#define XP_OFF    ((size_t)0)                         // f32 [2048][2048] = 16 MB
#define HSEQ_OFF  ((size_t)16777216)                  // f32 [2048][512]  = 4 MB
#define WT_OFF    ((size_t)20971520)                  // f32 [512][2048]  = 4 MB (W_hh^T)
#define FEATB_OFF ((size_t)25165824)                  // bf16 [2048][1024]= 4 MB
#define WIHB_OFF  ((size_t)29360128)                  // bf16 [2048][512] = 2 MB
#define AFFWB_OFF ((size_t)31457280)                  // bf16 [32000][1024] = 62.5 MB

// ---------------- helpers ----------------
__device__ __forceinline__ __bf16 f2bf(float f) {
  union { float f; unsigned u; } a; a.f = f;
  unsigned r = a.u + 0x7FFFu + ((a.u >> 16) & 1u);   // round-to-nearest-even
  union { unsigned short s; __bf16 b; } o; o.s = (unsigned short)(r >> 16);
  return o.b;
}

__device__ __forceinline__ f32x8 wmma_bf16(bf16x16 a, bf16x16 b, f32x8 c) {
  return __builtin_amdgcn_wmma_f32_16x16x32_bf16(false, a, false, b, (short)0, c,
                                                 false, false);
}

// A-fragment from a bf16 row-major row: needs row[k..k+15]
__device__ __forceinline__ bf16x16 load_a_bf16(const __bf16* p) {
  bf16x8 lo = *reinterpret_cast<const bf16x8*>(p);
  bf16x8 hi = *reinterpret_cast<const bf16x8*>(p + 8);
  return __builtin_shufflevector(lo, hi, 0,1,2,3,4,5,6,7,8,9,10,11,12,13,14,15);
}

// A-fragment gathered from an f32 row (embedding), converted to bf16 in-register
__device__ __forceinline__ bf16x16 load_a_f32(const float* p) {
  floatx8 lo = *reinterpret_cast<const floatx8*>(p);
  floatx8 hi = *reinterpret_cast<const floatx8*>(p + 8);
  bf16x16 a;
#pragma unroll
  for (int i = 0; i < 8; ++i) { a[i] = f2bf(lo[i]); a[i + 8] = f2bf(hi[i]); }
  return a;
}

// ---------------- prep kernels ----------------
__global__ void k_cvt_bf16(const float* __restrict__ src, __bf16* __restrict__ dst, int n) {
  int i = blockIdx.x * 256 + threadIdx.x;
  if (i < n) dst[i] = f2bf(src[i]);
}

__global__ void k_transpose_whh(const float* __restrict__ whh, float* __restrict__ wt) {
  int idx = blockIdx.x * 256 + threadIdx.x;          // idx = k*2048 + j
  if (idx < ND * G4H) {
    int k = idx >> 11, j = idx & 2047;
    wt[idx] = whh[j * ND + k];
  }
}

// ---------------- K1: xp = gather(embed_W, xs) @ W_ih^T + (b_ih+b_hh) ----------------
// M=2048, N=2048, K=512. bf16 WMMA, f32 accumulate. grid(8,32), block 256.
__global__ __launch_bounds__(256) void k_xp_gemm(
    const int* __restrict__ xs, const float* __restrict__ embW,
    const __bf16* __restrict__ wihb, const float* __restrict__ b_ih,
    const float* __restrict__ b_hh, float* __restrict__ xp) {
  const int lane = threadIdx.x & 31, wave = threadIdx.x >> 5;
  const int half = lane >> 4, l16 = lane & 15;
  const int mWave = blockIdx.y * 64 + (wave & 1) * 32;
  const int nWave = blockIdx.x * 256 + (wave >> 1) * 64;

  const float* pa0 = embW + (size_t)xs[mWave + l16] * ND;
  const float* pa1 = embW + (size_t)xs[mWave + 16 + l16] * ND;

  f32x8 acc[2][4];
#pragma unroll
  for (int mi = 0; mi < 2; ++mi)
#pragma unroll
    for (int ni = 0; ni < 4; ++ni) acc[mi][ni] = f32x8{};

  for (int k = 0; k < ND; k += 32) {
    const int kA = k + half * 16;
    bf16x16 a0 = load_a_f32(pa0 + kA);
    bf16x16 a1 = load_a_f32(pa1 + kA);
    const int kB = k + half * 16;
    bf16x16 bm[4];
#pragma unroll
    for (int ni = 0; ni < 4; ++ni)
      bm[ni] = *reinterpret_cast<const bf16x16*>(wihb + (nWave + ni * 16 + l16) * ND + kB);
#pragma unroll
    for (int ni = 0; ni < 4; ++ni) {
      acc[0][ni] = wmma_bf16(a0, bm[ni], acc[0][ni]);
      acc[1][ni] = wmma_bf16(a1, bm[ni], acc[1][ni]);
    }
  }
#pragma unroll
  for (int mi = 0; mi < 2; ++mi)
#pragma unroll
    for (int ni = 0; ni < 4; ++ni) {
      const int n = nWave + ni * 16 + l16;
      const float bias = b_ih[n] + b_hh[n];
      const int mBase = mWave + mi * 16 + half * 8;
#pragma unroll
      for (int r = 0; r < 8; ++r) xp[(mBase + r) * G4H + n] = acc[mi][ni][r] + bias;
    }
}

// ---------------- K2: LSTM recurrence, one WG per batch ----------------
// 1024 threads: hj = tid&511 , kq = tid>>9 (2-way K split, LDS reduce). f32.
__global__ __launch_bounds__(1024) void k_lstm(
    const float* __restrict__ xp, const float* __restrict__ wt,
    const float* __restrict__ h0, const float* __restrict__ c0,
    float* __restrict__ hseq, __bf16* __restrict__ featb,
    float* __restrict__ hT, float* __restrict__ cT) {
  const int b = blockIdx.x, tid = threadIdx.x;
  const int hj = tid & 511, kq = tid >> 9;
  __shared__ float hsh[NH];
  __shared__ float part[NH * 4];

  float c = (kq == 0) ? c0[b * NH + hj] : 0.f;
  if (tid < NH) hsh[tid] = h0[b * NH + tid];
  __syncthreads();

  const float* wtk = wt + (size_t)kq * 256 * G4H;
  const float* hb = hsh + kq * 256;
  for (int t = 0; t < TD; ++t) {
    float ai = 0.f, af = 0.f, ag = 0.f, ao = 0.f;
#pragma unroll 4
    for (int k = 0; k < 256; ++k) {
      const float hk = hb[k];
      const float* w = wtk + k * G4H;
      ai = fmaf(hk, w[hj], ai);
      af = fmaf(hk, w[512 + hj], af);
      ag = fmaf(hk, w[1024 + hj], ag);
      ao = fmaf(hk, w[1536 + hj], ao);
    }
    if (kq == 1) {
      part[hj * 4 + 0] = ai; part[hj * 4 + 1] = af;
      part[hj * 4 + 2] = ag; part[hj * 4 + 3] = ao;
    }
    __syncthreads();
    if (kq == 0) {
      const float* xr = xp + (size_t)(b * TD + t) * G4H;
      ai += part[hj * 4 + 0] + xr[hj];
      af += part[hj * 4 + 1] + xr[512 + hj];
      ag += part[hj * 4 + 2] + xr[1024 + hj];
      ao += part[hj * 4 + 3] + xr[1536 + hj];
      const float ig = 1.f / (1.f + __expf(-ai));
      const float fg = 1.f / (1.f + __expf(-af));
      const float gg = tanhf(ag);
      const float og = 1.f / (1.f + __expf(-ao));
      c = fg * c + ig * gg;
      const float h = og * tanhf(c);
      const int m = b * TD + t;
      hseq[(size_t)m * NH + hj] = h;
      featb[(size_t)m * KF + hj] = f2bf(h);
      hsh[hj] = h;
      if (t == TD - 1) { hT[b * NH + hj] = h; cT[b * NH + hj] = c; }
    }
    __syncthreads();
  }
}

// ---------------- K3: attention, one WG per batch ----------------
__global__ __launch_bounds__(256) void k_attn(
    const float* __restrict__ hs, const float* __restrict__ hseq,
    float* __restrict__ att_out, __bf16* __restrict__ featb) {
  const int b = blockIdx.x, tid = threadIdx.x;
  __shared__ float s[TE * TD]; // [e][t], 16 KB
  const float* hsb = hs + (size_t)b * TE * NH;
  const float* outb = hseq + (size_t)b * TD * NH;

  for (int idx = tid; idx < TE * TD; idx += 256) {
    const int e = idx >> 6, t = idx & 63;
    const float* pe = hsb + e * NH;
    const float* pt = outb + t * NH;
    float acc = 0.f;
    for (int k = 0; k < NH; ++k) acc = fmaf(pe[k], pt[k], acc);
    s[idx] = acc;
  }
  __syncthreads();
  if (tid < TD) { // softmax over e (column-wise); bank-conflict free (stride 64)
    const int t = tid;
    float mx = -1e30f;
    for (int e = 0; e < TE; ++e) mx = fmaxf(mx, s[e * 64 + t]);
    float sum = 0.f;
    for (int e = 0; e < TE; ++e) { float v = __expf(s[e * 64 + t] - mx); s[e * 64 + t] = v; sum += v; }
    const float inv = 1.f / sum;
    for (int e = 0; e < TE; ++e) {
      const float v = s[e * 64 + t] * inv;
      s[e * 64 + t] = v;
      att_out[(size_t)b * (TE * TD) + e * 64 + t] = v;
    }
  }
  __syncthreads();
  for (int idx = tid; idx < TD * NH; idx += 256) {
    const int t = idx >> 9, h = idx & 511;
    float acc = 0.f;
    for (int e = 0; e < TE; ++e) acc = fmaf(s[e * 64 + t], hsb[e * NH + h], acc);
    featb[(size_t)(b * TD + t) * KF + NH + h] = f2bf(acc); // context half of feat
  }
}

// ---------------- K4: logits = feat @ aff_W^T + aff_b ----------------
// M=2048, N=32000, K=1024. bf16 WMMA. grid(125,32), block 256.
__global__ __launch_bounds__(256) void k_logits_gemm(
    const __bf16* __restrict__ featb, const __bf16* __restrict__ affwb,
    const float* __restrict__ aff_b, float* __restrict__ out) {
  const int lane = threadIdx.x & 31, wave = threadIdx.x >> 5;
  const int half = lane >> 4, l16 = lane & 15;
  const int mWave = blockIdx.y * 64 + (wave & 1) * 32;
  const int nWave = blockIdx.x * 256 + (wave >> 1) * 64;

  const __bf16* pa0 = featb + (size_t)(mWave + l16) * KF;
  const __bf16* pa1 = featb + (size_t)(mWave + 16 + l16) * KF;

  f32x8 acc[2][4];
#pragma unroll
  for (int mi = 0; mi < 2; ++mi)
#pragma unroll
    for (int ni = 0; ni < 4; ++ni) acc[mi][ni] = f32x8{};

  for (int k = 0; k < KF; k += 32) {
    const int kA = k + half * 16;
    bf16x16 a0 = load_a_bf16(pa0 + kA);
    bf16x16 a1 = load_a_bf16(pa1 + kA);
    const int kB = k + half * 16;
    bf16x16 bm[4];
#pragma unroll
    for (int ni = 0; ni < 4; ++ni)
      bm[ni] = *reinterpret_cast<const bf16x16*>(affwb + (size_t)(nWave + ni * 16 + l16) * KF + kB);
#pragma unroll
    for (int ni = 0; ni < 4; ++ni) {
      acc[0][ni] = wmma_bf16(a0, bm[ni], acc[0][ni]);
      acc[1][ni] = wmma_bf16(a1, bm[ni], acc[1][ni]);
    }
  }
#pragma unroll
  for (int mi = 0; mi < 2; ++mi)
#pragma unroll
    for (int ni = 0; ni < 4; ++ni) {
      const int n = nWave + ni * 16 + l16;
      const float bias = aff_b[n];
      const int mBase = mWave + mi * 16 + half * 8;
#pragma unroll
      for (int r = 0; r < 8; ++r)
        out[(size_t)(mBase + r) * NV + n] = acc[mi][ni][r] + bias;
    }
}

// ---------------- launch ----------------
extern "C" void kernel_launch(void* const* d_in, const int* in_sizes, int n_in,
                              void* d_out, int out_size, void* d_ws, size_t ws_size,
                              hipStream_t stream) {
  const int*   xs     = (const int*)d_in[0];
  const float* hs     = (const float*)d_in[1];
  const float* h0     = (const float*)d_in[2];
  const float* c0     = (const float*)d_in[3];
  const float* embW   = (const float*)d_in[4];
  const float* W_ih   = (const float*)d_in[5];
  const float* W_hh   = (const float*)d_in[6];
  const float* b_ih   = (const float*)d_in[7];
  const float* b_hh   = (const float*)d_in[8];
  const float* aff_W  = (const float*)d_in[9];
  const float* aff_b  = (const float*)d_in[10];

  float*  out   = (float*)d_out;
  char*   ws    = (char*)d_ws;
  float*  xp    = (float*)(ws + XP_OFF);
  float*  hseq  = (float*)(ws + HSEQ_OFF);
  float*  wt    = (float*)(ws + WT_OFF);
  __bf16* featb = (__bf16*)(ws + FEATB_OFF);
  __bf16* wihb  = (__bf16*)(ws + WIHB_OFF);
  __bf16* affwb = (__bf16*)(ws + AFFWB_OFF);

  // prep: weight conversions/transposes (independent)
  {
    int n = G4H * ND;
    k_cvt_bf16<<<(n + 255) / 256, 256, 0, stream>>>(W_ih, wihb, n);
  }
  {
    int n = NV * KF;
    k_cvt_bf16<<<(n + 255) / 256, 256, 0, stream>>>(aff_W, affwb, n);
  }
  {
    int n = ND * G4H;
    k_transpose_whh<<<(n + 255) / 256, 256, 0, stream>>>(W_hh, wt);
  }

  // input projection (embedding gather fused into A-tile loads)
  k_xp_gemm<<<dim3(G4H / 256, NM / 64), 256, 0, stream>>>(xs, embW, wihb, b_ih, b_hh, xp);

  // LSTM recurrence (batch-parallel, no cross-WG sync needed)
  k_lstm<<<NB, 1024, 0, stream>>>(xp, wt, h0, c0, hseq, featb,
                                  out + HT_OFF, out + CT_OFF);

  // attention + context (writes second half of bf16 feat)
  k_attn<<<NB, 256, 0, stream>>>(hs, hseq, out + ATT_OFF, featb);

  // dominant GEMM: logits
  k_logits_gemm<<<dim3(NV / 256, NM / 64), 256, 0, stream>>>(featb, affwb, aff_b, out);
}
